// RegionLoss_22144851378183
// MI455X (gfx1250) — compile-verified, matrix operand order checked by reading, not run on hardware
//
#include <hip/hip_runtime.h>
#include <hip/hip_bf16.h>
#include <math.h>

// ---------------- problem constants (from reference) ----------------
#define NB_   32
#define NA_   5
#define NC_   8
#define NH_   32
#define NW_   64
#define NT_   50
#define CH_   (NH_ * NW_)          // 2048: stride between channels (floats)
#define SIL_THRESH_ 0.6f
#define OBJ_SCALE_  10.0f

#define CELLS_PER_BATCH (NA_ * NH_ * NW_)   // 10240
#define BLK_            256
#define BLKS_PER_BATCH  (CELLS_PER_BATCH / BLK_)   // 40
#define NPART_          (NB_ * BLKS_PER_BATCH)     // 1280 (multiple of 256)

__constant__ float c_aw[NA_] = {1.08f, 3.42f, 6.63f, 9.42f, 16.62f};
__constant__ float c_al[NA_] = {1.19f, 4.41f, 11.38f, 5.11f, 10.52f};

typedef __attribute__((ext_vector_type(16))) _Float16 v16h;
typedef __attribute__((ext_vector_type(8)))  float    v8f;

__device__ __forceinline__ float sigmoidf_(float v) {
    return 1.0f / (1.0f + __expf(-v));
}

// IoU on (cx, cy, w, h) boxes, exactly mirroring the reference formula.
__device__ __forceinline__ float iou_cxcywh(float ax, float ay, float aw, float ah,
                                            float bx, float by, float bw, float bh) {
    float mx = fminf(ax - aw * 0.5f, bx - bw * 0.5f);
    float Mx = fmaxf(ax + aw * 0.5f, bx + bw * 0.5f);
    float my = fminf(ay - ah * 0.5f, by - bh * 0.5f);
    float My = fmaxf(ay + ah * 0.5f, by + bh * 0.5f);
    float cw = aw + bw - (Mx - mx);
    float chh = ah + bh - (My - my);
    float inter = (cw > 0.0f && chh > 0.0f) ? cw * chh : 0.0f;
    float uni = aw * ah + bw * bh - inter;
    return inter / uni;
}

// -------------------------------------------------------------------
// Kernel 1: per-batch target preprocessing.
// Record per (b,t): [gx, gy, gw, gl, im, re, meta(int bits), tcls(int bits)]
// meta = gi | gj<<8 | best_n<<16 | keep<<20 | valid<<21
// -------------------------------------------------------------------
__global__ void __launch_bounds__(64)
prep_targets(const float* __restrict__ target, float* __restrict__ trec) {
    __shared__ int s_nz[NT_];
    __shared__ int s_val[NT_];
    __shared__ int s_cell[NT_];

    const int b = blockIdx.x;
    const int t = threadIdx.x;

    float gx = 0.f, gy = 0.f, gw = 0.f, gl = 0.f, im = 0.f, re = 0.f, cls = 0.f;

    if (t < NT_) {
        const float* tg = target + ((size_t)b * NT_ + t) * 7;
        cls = tg[0];
        float x1 = tg[1], y1 = tg[2], w1 = tg[3], l1 = tg[4];
        im = tg[5]; re = tg[6];
        s_nz[t] = (x1 != 0.0f) ? 1 : 0;
        gx = x1 * (float)NW_;
        gy = y1 * (float)NH_;
        gw = w1 * (float)NW_;
        gl = l1 * (float)NH_;
        int gi = min(max((int)gx, 0), NW_ - 1);
        int gj = min(max((int)gy, 0), NH_ - 1);
        // best anchor by anchor-space IoU (first max wins, like jnp.argmax)
        float best = -1.0e30f;
        int bn = 0;
        #pragma unroll
        for (int a = 0; a < NA_; ++a) {
            float aw = c_aw[a], al = c_al[a];
            float inter = fminf(gw, aw) * fminf(gl, al);
            float uni = gw * gl + aw * al - inter;
            float r = inter / uni;
            if (r > best) { best = r; bn = a; }
        }
        s_cell[t] = gi | (gj << 8) | (bn << 16);
    }
    __syncthreads();
    if (t < NT_) {
        int valid = 1;
        for (int u = 0; u <= t; ++u) valid &= s_nz[u];   // cumprod semantics
        s_val[t] = valid;
    }
    __syncthreads();
    if (t < NT_) {
        int valid = s_val[t];
        int keep = valid;
        if (keep) {
            const int mycell = s_cell[t];
            for (int u = t + 1; u < NT_; ++u) {          // highest index wins
                if (s_val[u] && s_cell[u] == mycell) { keep = 0; break; }
            }
        }
        int meta = s_cell[t] | (keep << 20) | (valid << 21);
        int tcls = min(max((int)cls, 0), NC_ - 1);
        float* r = trec + ((size_t)b * NT_ + t) * 8;
        r[0] = gx; r[1] = gy; r[2] = gw; r[3] = gl; r[4] = im; r[5] = re;
        ((int*)r)[6] = meta;
        ((int*)r)[7] = tcls;
    }
}

// -------------------------------------------------------------------
// Kernel 2: one thread per (b, a, h, w) cell; single streaming pass.
// -------------------------------------------------------------------
__global__ void __launch_bounds__(BLK_)
region_main(const float* __restrict__ outp, const float* __restrict__ trec,
            float* __restrict__ partial) {
    __shared__ float sgx[NT_], sgy[NT_], sgw[NT_], sgl[NT_], sim[NT_], sre[NT_];
    __shared__ int   smeta[NT_], scls[NT_];
    __shared__ float red[BLK_];

    const int b = blockIdx.y;

    for (int t = threadIdx.x; t < NT_; t += BLK_) {
        const float* r = trec + ((size_t)b * NT_ + t) * 8;
        sgx[t] = r[0]; sgy[t] = r[1]; sgw[t] = r[2]; sgl[t] = r[3];
        sim[t] = r[4]; sre[t] = r[5];
        smeta[t] = ((const int*)r)[6];
        scls[t]  = ((const int*)r)[7];
    }
    __syncthreads();

    const int cell = blockIdx.x * BLK_ + threadIdx.x;      // 0..10239
    const int a  = cell / CH_;
    const int hw = cell % CH_;
    const int h  = hw / NW_;
    const int wc = hw % NW_;

    const float* base = outp + ((size_t)(b * (NA_ * 15) + a * 15)) * CH_ + hw;
    const float o0 = base[0 * CH_];
    const float o1 = base[1 * CH_];
    const float o2 = base[2 * CH_];
    const float o3 = base[3 * CH_];
    const float o4 = base[4 * CH_];
    const float o5 = base[5 * CH_];
    const float o6 = base[6 * CH_];

    const float px = sigmoidf_(o0);
    const float py = sigmoidf_(o1);
    const float pconf = sigmoidf_(o6);
    const float pbx = px + (float)wc;
    const float pby = py + (float)h;
    const float pbw = __expf(o2) * c_aw[a];
    const float pbl = __expf(o3) * c_al[a];

    const int mykey = wc | (h << 8) | (a << 16);
    float max_iou = 0.0f;
    int match = -1;
    for (int t = 0; t < NT_; ++t) {
        const int meta = smeta[t];
        if (!((meta >> 21) & 1)) continue;                 // invalid target
        float iou = iou_cxcywh(sgx[t], sgy[t], sgw[t], sgl[t], pbx, pby, pbw, pbl);
        max_iou = fmaxf(max_iou, iou);
        if ((meta & (1 << 20)) && (meta & 0xFFFFF) == mykey) match = t;
    }

    float loss = 0.0f;
    if (match >= 0) {
        const int meta = smeta[match];
        const int gi = meta & 0xFF, gj = (meta >> 8) & 0xFF;
        const float gx = sgx[match], gy = sgy[match];
        const float gw = sgw[match], gl = sgl[match];
        const float tx = gx - (float)gi;
        const float ty = gy - (float)gj;
        const float tw = __logf(gw / c_aw[a]);
        const float tl = __logf(gl / c_al[a]);
        float d;
        d = px - tx;          loss += d * d;
        d = py - ty;          loss += d * d;
        d = o2 - tw;          loss += d * d;
        d = o3 - tl;          loss += d * d;
        d = o4 - sim[match];  loss += d * d;
        d = o5 - sre[match];  loss += d * d;
        const float tconf = iou_cxcywh(gx, gy, gw, gl, pbx, pby, pbw, pbl);
        d = pconf - tconf;
        loss += (OBJ_SCALE_ * OBJ_SCALE_) * d * d;
        // class cross-entropy (logits read only for matched cells)
        const float* cb = base + 7 * CH_;
        float lg[NC_];
        float m = -1.0e30f;
        #pragma unroll
        for (int c = 0; c < NC_; ++c) { lg[c] = cb[c * CH_]; m = fmaxf(m, lg[c]); }
        float s = 0.0f;
        #pragma unroll
        for (int c = 0; c < NC_; ++c) s += __expf(lg[c] - m);
        loss += (m + __logf(s)) - lg[scls[match]];
    } else {
        const float mask = (max_iou > SIL_THRESH_) ? 0.0f : 1.0f;
        loss += mask * pconf * pconf;                      // tconf == 0 here
    }

    // block tree reduction
    red[threadIdx.x] = loss;
    __syncthreads();
    for (int off = BLK_ / 2; off > 0; off >>= 1) {
        if (threadIdx.x < off) red[threadIdx.x] += red[threadIdx.x + off];
        __syncthreads();
    }
    if (threadIdx.x == 0)
        partial[blockIdx.y * BLKS_PER_BATCH + blockIdx.x] = red[0];
}

// -------------------------------------------------------------------
// Kernel 3: single-wave WMMA reduction of the 1280 partials.
// Each f32 partial is split p = hi + lo (f16 halves); A carries values,
// B is all-ones, C (f32) accumulates across issues: D[m][n] = sum_k A[m][k].
// Column n==0 lives in lanes 0 (rows 0..7) and 16 (rows 8..15).
// -------------------------------------------------------------------
__global__ void __launch_bounds__(32)
wmma_reduce(const float* __restrict__ partial, float* __restrict__ outp) {
    const int lane = threadIdx.x;
    v8f c = {};
    v16h ones;
    #pragma unroll
    for (int i = 0; i < 16; ++i) ones[i] = (_Float16)1.0f;

    #pragma unroll
    for (int it = 0; it < NPART_ / 256; ++it) {          // 5 WMMA issues
        v16h av;
        const int basei = it * 256 + lane * 8;
        #pragma unroll
        for (int k = 0; k < 8; ++k) {
            float p = partial[basei + k];
            _Float16 hi = (_Float16)p;
            float r = p - (float)hi;                     // exact residual
            av[2 * k]     = hi;
            av[2 * k + 1] = (_Float16)r;
        }
        c = __builtin_amdgcn_wmma_f32_16x16x32_f16(
                /*neg_a=*/false, av, /*neg_b=*/false, ones,
                /*c_mod=*/(short)0, c, /*reuse_a=*/false, /*reuse_b=*/false);
    }
    float s = c[0] + c[1] + c[2] + c[3] + c[4] + c[5] + c[6] + c[7];
    float s_other = __shfl(s, lane ^ 16, 32);            // pair lanes 0 <-> 16
    if (lane == 0) outp[0] = s + s_other;
}

// -------------------------------------------------------------------
extern "C" void kernel_launch(void* const* d_in, const int* in_sizes, int n_in,
                              void* d_out, int out_size, void* d_ws, size_t ws_size,
                              hipStream_t stream) {
    (void)in_sizes; (void)n_in; (void)out_size; (void)ws_size;
    const float* output = (const float*)d_in[0];
    const float* target = (const float*)d_in[1];
    float* ws      = (float*)d_ws;
    float* trec    = ws;                        // NB_*NT_*8 floats = 12800
    float* partial = ws + (size_t)NB_ * NT_ * 8; // NPART_ floats = 1280

    prep_targets<<<NB_, 64, 0, stream>>>(target, trec);
    dim3 grid(BLKS_PER_BATCH, NB_);
    region_main<<<grid, BLK_, 0, stream>>>(output, trec, partial);
    wmma_reduce<<<1, 32, 0, stream>>>(partial, (float*)d_out);
}
